// AMMP_65438121722509
// MI455X (gfx1250) — matrix operand order repaired and checked by reading, not compiled, not run on hardware
//
#include <hip/hip_runtime.h>
#include <hip/hip_bf16.h>

typedef _Float16 h16v __attribute__((ext_vector_type(16)));
typedef _Float16 h8v  __attribute__((ext_vector_type(8)));
typedef float    f8v  __attribute__((ext_vector_type(8)));
typedef float    f4v  __attribute__((ext_vector_type(4)));

#define HW   65536
#define CCH  64
#define EPSV 1e-5f
#define KSEL 26214u   // int(0.4 * 65536)

// Branch-free exact-GELU: erf via Abramowitz-Stegun 7.1.26 (|err|<=1.5e-7),
// one v_exp_f32 + one v_rcp_f32 + FMAs, no EXEC divergence.
__device__ __forceinline__ float gelu_f(float x) {
    float z  = x * 0.70710678118654752f;
    float az = __builtin_fabsf(z);
    float t  = __builtin_amdgcn_rcpf(__builtin_fmaf(0.3275911f, az, 1.0f));
    float p  = t * (0.254829592f + t * (-0.284496736f +
               t * (1.421413741f + t * (-1.453152027f + t * 1.061405429f))));
    float e  = __builtin_amdgcn_exp2f(az * az * -1.4426950408889634f);
    float er = __builtin_copysignf(1.0f - p * e, z);
    return 0.5f * x * (1.0f + er);
}

// Cross-half (lane ^ 16) exchange: single ds_swizzle (SWAPX16), no bpermute fixups.
// group-of-32 encoding: and=0x1F, or=0, xor=0x10 -> 0x401F
__device__ __forceinline__ float swap16_f(float v) {
    return __int_as_float(__builtin_amdgcn_ds_swizzle(__float_as_int(v), 0x401F));
}

// A-matrix gather (16x32 f16): row-major LDS table [M][K], rowstride rs halves.
__device__ __forceinline__ h16v lds_tileA(const _Float16* tab, int rs, int mbase, int kbase, int lane) {
    int row = mbase + (lane & 15);
    int o   = (lane & 16) ? 8 : 0;
    const _Float16* p = tab + row * rs + kbase + o;
    h8v lo = *(const h8v*)(p);
    h8v hi = *(const h8v*)(p + 16);
    return __builtin_shufflevector(lo, hi, 0,1,2,3,4,5,6,7,8,9,10,11,12,13,14,15);
}

// B-matrix gather (32x16 f16): token-major LDS buffer [N][K], rowstride rs halves.
__device__ __forceinline__ h16v lds_tileB(const _Float16* buf, int rs, int tokbase, int kbase, int lane) {
    int tok = tokbase + (lane & 15);
    int o   = (lane & 16) ? 8 : 0;
    const _Float16* p = buf + tok * rs + kbase + o;
    h8v lo = *(const h8v*)(p);
    h8v hi = *(const h8v*)(p + 16);
    return __builtin_shufflevector(lo, hi, 0,1,2,3,4,5,6,7,8,9,10,11,12,13,14,15);
}

__device__ __forceinline__ unsigned ordkey(float f) {
    unsigned u = __float_as_uint(f);
    return (u & 0x80000000u) ? ~u : (u | 0x80000000u);
}

// ---------------- Pass 1: LN -> WMMA proj -> GELU -> per-channel partial sums ---------
__global__ __launch_bounds__(256) void k_pass1(const float* __restrict__ x,
                                               const float* __restrict__ ln_w,
                                               const float* __restrict__ ln_b,
                                               const float* __restrict__ proj_w,
                                               float* __restrict__ partials) {
    __shared__ __align__(16) _Float16 xnb[128 * 72];
    __shared__ __align__(16) _Float16 wA[64 * 64];
    __shared__ __align__(16) float    lsum[128 * 72];   // f32 local_x tile [tok][ch]
    __shared__ float lnw[64], lnb[64];
    int tid = threadIdx.x;
    int b   = blockIdx.x >> 9;
    int t0  = (blockIdx.x & 511) << 7;
    const float* xb = x + (size_t)b * CCH * HW + t0;
    if (tid < 64) __builtin_prefetch(xb + (size_t)tid * HW, 0, 1);  // global_prefetch_b8
    for (int i = tid; i < 4096; i += 256) wA[i] = (_Float16)proj_w[i];
    if (tid < 64) { lnw[tid] = ln_w[tid]; lnb[tid] = ln_b[tid]; }
    __syncthreads();
    if (tid < 128) {
        float s = 0.f, q = 0.f;
        for (int c = 0; c < 64; ++c) { float v = xb[(size_t)c * HW + tid]; s += v; q += v * v; }
        float mu = s * (1.f / 64.f);
        float var = q * (1.f / 64.f) - mu * mu;
        float ri = rsqrtf(var + EPSV);
        for (int c = 0; c < 64; ++c) {
            float v = xb[(size_t)c * HW + tid];
            xnb[tid * 72 + c] = (_Float16)(((v - mu) * ri) * lnw[c] + lnb[c]);
        }
    }
    __syncthreads();
    int lane = tid & 31, wid = tid >> 5;
    int tokbase = wid * 16;
    int off8 = (lane & 16) ? 8 : 0;
    int tok  = tokbase + (lane & 15);
#pragma unroll
    for (int mt = 0; mt < 4; ++mt) {
        int mbase = mt * 16;
        f8v d = {0.f, 0.f, 0.f, 0.f, 0.f, 0.f, 0.f, 0.f};
#pragma unroll
        for (int ks = 0; ks < 2; ++ks) {
            h16v a  = lds_tileA(wA, 64, mbase, ks * 32, lane);
            h16v bm = lds_tileB(xnb, 72, tokbase, ks * 32, lane);
            d = __builtin_amdgcn_wmma_f32_16x16x32_f16(false, a, false, bm, (short)0, d, false, false);
        }
        f4v g0 = {gelu_f(d[0]), gelu_f(d[1]), gelu_f(d[2]), gelu_f(d[3])};
        f4v g1 = {gelu_f(d[4]), gelu_f(d[5]), gelu_f(d[6]), gelu_f(d[7])};
        *(f4v*)(&lsum[tok * 72 + mbase + off8])     = g0;   // ds_store_b128
        *(f4v*)(&lsum[tok * 72 + mbase + off8 + 4]) = g1;
    }
    __syncthreads();
    if (tid < 64) {                     // fixed-order (deterministic) token reduction
        float s = 0.f;
        for (int t = 0; t < 128; ++t) s += lsum[t * 72 + tid];
        partials[(size_t)blockIdx.x * 64 + tid] = s;
    }
}

// ---------------- Per-batch constants: global_x, scale, fused W1, bias, misc ----------
__global__ __launch_bounds__(256) void k_const(const float* __restrict__ partials,
                                               const float* __restrict__ ratio,
                                               const float* __restrict__ ctrl_w,
                                               const float* __restrict__ cc_w1,
                                               const float* __restrict__ chan_w,
                                               const float* __restrict__ comp3_w,
                                               const float* __restrict__ alpha,
                                               float* __restrict__ gx, float* __restrict__ gb,
                                               _Float16* __restrict__ w1s, float* __restrict__ misc,
                                               unsigned* __restrict__ state, unsigned* __restrict__ hist) {
    __shared__ float scaleL[1024];
    __shared__ float gxL[512];
    int tid = threadIdx.x;
    for (int i = tid; i < 512; i += 256) {            // global_x = spatial mean
        int b = i >> 6, c = i & 63;
        float s = 0.f;
        const float* p = partials + ((size_t)b * 512) * 64 + c;
        for (int j = 0; j < 512; ++j) s += p[(size_t)j * 64];
        float g = s * (1.f / 65536.f);
        gxL[i] = g; gx[i] = g;
    }
    for (int i = tid; i < 1024; i += 256) {           // scale = ratio @ exp(ctrl_w).T
        int b = i >> 7, o = i & 127;
        float s = 0.f;
        for (int k = 0; k < 6; ++k) s += ratio[b * 6 + k] * expf(ctrl_w[o * 6 + k]);
        scaleL[i] = s;
    }
    __syncthreads();
    for (int i = tid; i < 32768; i += 256) {          // W1s[b][o][c] = cc_w1[o][c]*scale[b][c]
        int b = i >> 12, o = (i >> 6) & 63, c = i & 63;
        w1s[i] = (_Float16)(cc_w1[o * 128 + c] * scaleL[b * 128 + c]);
    }
    for (int i = tid; i < 512; i += 256) {            // bias from global branch
        int b = i >> 6, o = i & 63;
        float s = 0.f;
        for (int c = 0; c < 64; ++c)
            s += cc_w1[o * 128 + 64 + c] * gxL[b * 64 + c] * scaleL[b * 128 + 64 + c];
        gb[i] = s;
    }
    for (int i = tid; i < 4096; i += 256) hist[i] = 0u;
    if (tid < 8) {
        state[tid * 4 + 0] = 0u; state[tid * 4 + 1] = KSEL;
        state[tid * 4 + 2] = 0u; state[tid * 4 + 3] = KSEL;
    }
    if (tid == 0) {
        misc[0] = chan_w[0]; misc[1] = chan_w[1]; misc[2] = chan_w[2]; misc[3] = chan_w[3];
        misc[4] = comp3_w[0]; misc[5] = comp3_w[1]; misc[6] = comp3_w[2]; misc[7] = comp3_w[3];
        float am = alpha[0];
        for (int k = 0; k < 48; ++k) am = fmaxf(am, ratio[k]);
        misc[8] = 1.f / (1.f + expf(-am));
    }
}

// ---------------- Pass 2: full fused pipeline -> scores + weighting ------------------
__global__ __launch_bounds__(256) void k_pass2(const float* __restrict__ x,
                                               const float* __restrict__ ln_w,
                                               const float* __restrict__ ln_b,
                                               const float* __restrict__ proj_w,
                                               const float* __restrict__ cc_w2,
                                               const float* __restrict__ cc_w3,
                                               const float* __restrict__ gx,
                                               const float* __restrict__ gb,
                                               const _Float16* __restrict__ w1s,
                                               const float* __restrict__ misc,
                                               float* __restrict__ scores,
                                               float* __restrict__ outw) {
    __shared__ __align__(16) _Float16 xnb[128 * 72];   // xn, later y1
    __shared__ __align__(16) _Float16 locb[128 * 72];  // local_x, later y2
    __shared__ __align__(16) _Float16 wA[4096];
    __shared__ __align__(16) _Float16 w1[4096];
    __shared__ __align__(16) _Float16 w2[2048];
    __shared__ float w3f[64], gxl[64], gbl[64], lnw[64], lnb[64], miscl[16];
    int tid = threadIdx.x;
    int b   = blockIdx.x >> 9;
    int t0  = (blockIdx.x & 511) << 7;
    const float* xb = x + (size_t)b * CCH * HW + t0;
    if (tid < 64) __builtin_prefetch(xb + (size_t)tid * HW, 0, 1);  // global_prefetch_b8
    for (int i = tid; i < 4096; i += 256) { wA[i] = (_Float16)proj_w[i]; w1[i] = w1s[(size_t)b * 4096 + i]; }
    for (int i = tid; i < 2048; i += 256) w2[i] = (_Float16)cc_w2[i];
    if (tid < 64) {
        w3f[tid] = cc_w3[tid]; gxl[tid] = gx[b * 64 + tid]; gbl[tid] = gb[b * 64 + tid];
        lnw[tid] = ln_w[tid];  lnb[tid] = ln_b[tid];
    }
    if (tid < 16) miscl[tid] = (tid < 9) ? misc[tid] : 0.f;
    __syncthreads();
    if (tid < 128) {
        float s = 0.f, q = 0.f;
        for (int c = 0; c < 64; ++c) { float v = xb[(size_t)c * HW + tid]; s += v; q += v * v; }
        float mu = s * (1.f / 64.f);
        float var = q * (1.f / 64.f) - mu * mu;
        float ri = rsqrtf(var + EPSV);
        for (int c = 0; c < 64; ++c) {
            float v = xb[(size_t)c * HW + tid];
            xnb[tid * 72 + c] = (_Float16)(((v - mu) * ri) * lnw[c] + lnb[c]);
        }
    }
    __syncthreads();
    int lane = tid & 31, wid = tid >> 5;
    int tokbase = wid * 16;
    int off8 = (lane & 16) ? 8 : 0;
    int tok = tokbase + (lane & 15);
    // ---- proj + GELU + pooled stats ----
    float pm = -1e30f, ps = 0.f;
#pragma unroll
    for (int mt = 0; mt < 4; ++mt) {
        int mbase = mt * 16;
        f8v d = {0.f, 0.f, 0.f, 0.f, 0.f, 0.f, 0.f, 0.f};
#pragma unroll
        for (int ks = 0; ks < 2; ++ks) {
            h16v a  = lds_tileA(wA, 64, mbase, ks * 32, lane);
            h16v bm = lds_tileB(xnb, 72, tokbase, ks * 32, lane);
            d = __builtin_amdgcn_wmma_f32_16x16x32_f16(false, a, false, bm, (short)0, d, false, false);
        }
#pragma unroll
        for (int v = 0; v < 8; ++v) {
            int m = mbase + v + off8;
            float g = gelu_f(d[v]);
            locb[tok * 72 + m] = (_Float16)g;
            float gv = g * gxl[m];
            pm = fmaxf(pm, gv); ps += gv;
        }
    }
    pm = fmaxf(pm, swap16_f(pm));
    ps = ps + swap16_f(ps);
    float mean = ps * (1.f / 64.f);
    float x1_0 = gelu_f(miscl[0] * pm + miscl[1] * mean);
    float x1_1 = gelu_f(miscl[2] * pm + miscl[3] * mean);
    // ---- cc1 (scale-folded weights + global-branch bias) ----
#pragma unroll
    for (int mt = 0; mt < 4; ++mt) {
        int mbase = mt * 16;
        f8v d = {0.f, 0.f, 0.f, 0.f, 0.f, 0.f, 0.f, 0.f};
#pragma unroll
        for (int ks = 0; ks < 2; ++ks) {
            h16v a  = lds_tileA(w1, 64, mbase, ks * 32, lane);
            h16v bm = lds_tileB(locb, 72, tokbase, ks * 32, lane);
            d = __builtin_amdgcn_wmma_f32_16x16x32_f16(false, a, false, bm, (short)0, d, false, false);
        }
#pragma unroll
        for (int v = 0; v < 8; ++v) {
            int m = mbase + v + off8;
            xnb[tok * 72 + m] = (_Float16)gelu_f(d[v] + gbl[m]);
        }
    }
    // ---- cc2 ----
#pragma unroll
    for (int mt = 0; mt < 2; ++mt) {
        int mbase = mt * 16;
        f8v d = {0.f, 0.f, 0.f, 0.f, 0.f, 0.f, 0.f, 0.f};
#pragma unroll
        for (int ks = 0; ks < 2; ++ks) {
            h16v a  = lds_tileA(w2, 64, mbase, ks * 32, lane);
            h16v bm = lds_tileB(xnb, 72, tokbase, ks * 32, lane);
            d = __builtin_amdgcn_wmma_f32_16x16x32_f16(false, a, false, bm, (short)0, d, false, false);
        }
#pragma unroll
        for (int v = 0; v < 8; ++v) {
            int m = mbase + v + off8;
            locb[tok * 72 + m] = (_Float16)gelu_f(d[v]);
        }
    }
    // ---- cc3 (2x32) + final combine ----
    {
        int o = (lane >> 4) & 1;
        float acc = 0.f;
#pragma unroll
        for (int c = 0; c < 32; ++c) acc += (float)locb[tok * 72 + c] * w3f[o * 32 + c];
        float y3  = gelu_f(acc);
        float y3o = swap16_f(y3);
        if (lane < 16) {
            float sfin = miscl[4] * x1_0 + miscl[5] * x1_1 + miscl[6] * y3 + miscl[7] * y3o;
            float xo = gelu_f(sfin);
            size_t idx = (size_t)b * HW + t0 + tok;
            scores[idx] = xo;
            outw[idx]   = miscl[8] * gelu_f(xo);
        }
    }
}

// ---------------- Radix top-K / bottom-K selection -----------------------------------
__global__ __launch_bounds__(256) void k_hist(const float* __restrict__ scores,
                                              const unsigned* __restrict__ state,
                                              unsigned* __restrict__ hist, int p) {
    __shared__ unsigned hh[256], hl[256];
    int tid = threadIdx.x;
    hh[tid] = 0u; hl[tid] = 0u;
    __syncthreads();
    size_t i = (size_t)blockIdx.x * 256 + tid;
    int b = (int)(i >> 16);
    unsigned key = ordkey(scores[i]);
    unsigned uhi = ~key, ulo = key;
    int shift = 24 - 8 * p;
    unsigned phi = state[b * 4 + 0], plo = state[b * 4 + 2];
    if (p == 0 || (uhi >> (shift + 8)) == phi) atomicAdd(&hh[(uhi >> shift) & 255u], 1u);
    if (p == 0 || (ulo >> (shift + 8)) == plo) atomicAdd(&hl[(ulo >> shift) & 255u], 1u);
    __syncthreads();
    if (hh[tid]) atomicAdd(&hist[b * 256 + tid], hh[tid]);
    if (hl[tid]) atomicAdd(&hist[(8 + b) * 256 + tid], hl[tid]);
}

__global__ void k_scan(unsigned* __restrict__ state, unsigned* __restrict__ hist) {
    int t = threadIdx.x;
    if (t < 16) {
        int b = t >> 1, which = t & 1;
        unsigned* h = &hist[(which ? (8 + b) : b) * 256];
        unsigned rem = state[b * 4 + (which ? 3 : 1)];
        unsigned pre = state[b * 4 + (which ? 2 : 0)];
        unsigned cum = 0, sel = 0; bool found = false;
        for (int bin = 0; bin < 256; ++bin) {
            unsigned c = h[bin]; h[bin] = 0u;
            if (!found) {
                if (cum + c >= rem) { sel = (unsigned)bin; found = true; }
                else cum += c;
            }
        }
        state[b * 4 + (which ? 2 : 0)] = (pre << 8) | sel;
        state[b * 4 + (which ? 3 : 1)] = rem - cum;
    }
}

__global__ __launch_bounds__(256) void k_mark(const float* __restrict__ scores,
                                              const unsigned* __restrict__ state,
                                              float* __restrict__ outm) {
    size_t i = (size_t)blockIdx.x * 256 + threadIdx.x;
    int b = (int)(i >> 16);
    unsigned key = ordkey(scores[i]);
    unsigned phi = state[b * 4 + 0], plo = state[b * 4 + 2];
    outm[i] = (((~key) <= phi) || (key <= plo)) ? 1.0f : 0.0f;
}

extern "C" void kernel_launch(void* const* d_in, const int* in_sizes, int n_in,
                              void* d_out, int out_size, void* d_ws, size_t ws_size,
                              hipStream_t stream) {
    (void)in_sizes; (void)n_in; (void)out_size; (void)ws_size;
    const float* x      = (const float*)d_in[0];
    const float* ratio  = (const float*)d_in[2];
    const float* ln_w   = (const float*)d_in[3];
    const float* ln_b   = (const float*)d_in[4];
    const float* ctrl_w = (const float*)d_in[5];
    const float* proj_w = (const float*)d_in[6];
    const float* chan_w = (const float*)d_in[7];
    const float* cc_w1  = (const float*)d_in[8];
    const float* cc_w2  = (const float*)d_in[9];
    const float* cc_w3  = (const float*)d_in[10];
    const float* comp3  = (const float*)d_in[11];
    const float* alpha  = (const float*)d_in[12];
    float* out = (float*)d_out;

    float*     ws_part   = (float*)d_ws;                       // 4096*64 f
    float*     ws_gx     = ws_part + 4096 * 64;                // 512 f
    float*     ws_gb     = ws_gx + 512;                        // 512 f
    _Float16*  ws_w1     = (_Float16*)(ws_gb + 512);           // 32768 h
    float*     ws_misc   = (float*)(ws_w1 + 32768);            // 16 f
    float*     ws_scores = ws_misc + 16;                       // 524288 f
    unsigned*  ws_state  = (unsigned*)(ws_scores + 524288);    // 32 u
    unsigned*  ws_hist   = ws_state + 32;                      // 4096 u

    k_pass1<<<4096, 256, 0, stream>>>(x, ln_w, ln_b, proj_w, ws_part);
    k_const<<<1, 256, 0, stream>>>(ws_part, ratio, ctrl_w, cc_w1, chan_w, comp3, alpha,
                                   ws_gx, ws_gb, ws_w1, ws_misc, ws_state, ws_hist);
    k_pass2<<<4096, 256, 0, stream>>>(x, ln_w, ln_b, proj_w, cc_w2, cc_w3,
                                      ws_gx, ws_gb, ws_w1, ws_misc, ws_scores, out + 524288);
    for (int p = 0; p < 4; ++p) {
        k_hist<<<2048, 256, 0, stream>>>(ws_scores, ws_state, ws_hist, p);
        k_scan<<<1, 64, 0, stream>>>(ws_state, ws_hist);
    }
    k_mark<<<2048, 256, 0, stream>>>(ws_scores, ws_state, out);
}
